// TemporalGraphNetwork_17910013624784
// MI455X (gfx1250) — compile-verified, matrix operand order checked by reading, not run on hardware
//
#include <hip/hip_runtime.h>
#include <hip/hip_bf16.h>

// ---------------------------------------------------------------------------
// Types for CDNA5 WMMA (wave32): A/B = 16x32 bf16 (16 elems/lane), C/D = v8f
// ---------------------------------------------------------------------------
typedef __attribute__((ext_vector_type(16))) __bf16          v16bf;
typedef __attribute__((ext_vector_type(8)))  float           v8f;
typedef __attribute__((ext_vector_type(4)))  unsigned int    u32x4;

struct Frag32B { u32x4 lo, hi; };   // 32 bytes = one 16-elem bf16 fragment

__device__ __forceinline__ unsigned short f2bf(float f) {
    unsigned int u = __float_as_uint(f);
    unsigned int r = u + 0x7FFFu + ((u >> 16) & 1u);   // round-to-nearest-even
    return (unsigned short)(r >> 16);
}

// Fragment k-pattern (16-bit 16x32 A, and B^T by symmetry):
// lane l, elems 0..7  -> k = (l>>4)*8 + 0..7      (contiguous)
// lane l, elems 8..15 -> k = 16 + (l>>4)*8 + 0..7 (contiguous)
// => fragment = two ds_load_b128 from k-contiguous storage.

// ---------------------------------------------------------------------------
// Generic bf16 WMMA GEMM:  C[M x NOUT] = [A1 | A2] @ W + bias
//   A1: M x K1 bf16 row-major, A2: M x K2 bf16 row-major
//   K1, K2 multiples of 32  (so each 32-k chunk has a single source)
//   W : packed bf16, layout [kChunk][col][32 klocal]
// Both operands staged global->LDS with async b128 DMA, double-buffered,
// prefetch of chunk c+1 overlapped with WMMA compute on chunk c.
// Workgroup: 256 threads = 8 waves; block = 32 rows x NOUT cols.
// ---------------------------------------------------------------------------
template <int TILES>
__global__ __launch_bounds__(256)
void gemm_bf16(const unsigned short* __restrict__ A1, int K1,
               const unsigned short* __restrict__ A2, int K2,
               const unsigned short* __restrict__ W,
               const float* __restrict__ bias,
               float* __restrict__ C, int M) {
    constexpr int NOUT = TILES * 64;
    constexpr int COLS_PER_WAVE = TILES * 16;

    __shared__ unsigned short As[2][32 * 32];     // [row][k] row-major
    __shared__ unsigned short Ws[2][NOUT * 32];   // [col][klocal]

    const int tid  = threadIdx.x;
    const int wave = tid >> 5;
    const int lane = tid & 31;
    const int laneHalf = lane >> 4;
    const int lane16   = lane & 15;

    const int rowBase = blockIdx.x * 32;
    const int wr = wave & 1;
    const int wc = wave >> 1;
    const int K  = K1 + K2;
    const int NC = K >> 5;

    auto prefetch = [&](int c, int buf) {
        // W chunk: NOUT*32 bf16 contiguous = TILES*256 lane-transfers of 16B.
        // Trip count is uniform & compile-time: no exec masking needed.
        const unsigned short* Wg = W + (size_t)c * NOUT * 32;
#pragma unroll
        for (int it = 0; it < TILES; ++it) {
            const int i = tid + it * 256;
            unsigned lds = (unsigned)(size_t)(&Ws[buf][0] + i * 8);
            unsigned long long g = (unsigned long long)(size_t)(Wg + i * 8);
            asm volatile("global_load_async_to_lds_b128 %0, %1, off"
                         :: "v"(lds), "v"(g) : "memory");
        }
        // A chunk: 32 rows x 64B (single source since K1 % 32 == 0)
        const int kc = c * 32;
        const unsigned short* Ag;
        int kloc, kstride;
        if (kc < K1) { Ag = A1; kloc = kc;      kstride = K1; }
        else         { Ag = A2; kloc = kc - K1; kstride = K2; }
        if (tid < 128) {
            int r = tid >> 2, part = tid & 3;
            int row = rowBase + r; if (row >= M) row = M - 1;
            unsigned lds = (unsigned)(size_t)(&As[buf][0] + r * 32 + part * 8);
            unsigned long long g =
                (unsigned long long)(size_t)(Ag + (size_t)row * kstride + kloc + part * 8);
            asm volatile("global_load_async_to_lds_b128 %0, %1, off"
                         :: "v"(lds), "v"(g) : "memory");
        }
    };

    v8f acc[TILES];
    const v8f zero = {0.f, 0.f, 0.f, 0.f, 0.f, 0.f, 0.f, 0.f};
#pragma unroll
    for (int t = 0; t < TILES; ++t) acc[t] = zero;

    prefetch(0, 0);
    for (int c = 0; c < NC; ++c) {
        const int buf = c & 1;
        asm volatile("s_wait_asynccnt 0x0" ::: "memory");  // my DMA done
        __syncthreads();                                   // everyone's DMA visible
        if (c + 1 < NC) prefetch(c + 1, buf ^ 1);          // overlap next chunk

        const unsigned short* ap = &As[buf][0] + (wr * 16 + lane16) * 32 + laneHalf * 8;
        Frag32B fa;
        fa.lo = *(const u32x4*)(ap);
        fa.hi = *(const u32x4*)(ap + 16);
        v16bf afrag = __builtin_bit_cast(v16bf, fa);

#pragma unroll
        for (int t = 0; t < TILES; ++t) {
            const int col = wc * COLS_PER_WAVE + t * 16 + lane16;
            const unsigned short* bp = &Ws[buf][0] + (size_t)col * 32 + laneHalf * 8;
            Frag32B fb;
            fb.lo = *(const u32x4*)(bp);
            fb.hi = *(const u32x4*)(bp + 16);
            v16bf bfrag = __builtin_bit_cast(v16bf, fb);
            acc[t] = __builtin_amdgcn_wmma_f32_16x16x32_bf16(
                false, afrag, false, bfrag, (short)0, acc[t], false, false);
        }
    }

    // C/D layout: lane l -> col = l&15; vgpr m -> row = (l>>4)*8 + m
#pragma unroll
    for (int t = 0; t < TILES; ++t) {
        const int col = wc * COLS_PER_WAVE + t * 16 + lane16;
        const float b = bias[col];
#pragma unroll
        for (int m = 0; m < 8; ++m) {
            const int row = rowBase + wr * 16 + laneHalf * 8 + m;
            if (row < M) C[(size_t)row * NOUT + col] = acc[t][m] + b;
        }
    }
}

// ---------------------------------------------------------------------------
// Prep kernels
// ---------------------------------------------------------------------------
__global__ void time_encode_k(const float* __restrict__ relt,
                              const float* __restrict__ tw,
                              const float* __restrict__ tb,
                              unsigned short* __restrict__ tenc, int E) {
    int i = blockIdx.x * blockDim.x + threadIdx.x;
    if (i < E * 64) {
        int e = i >> 6, j = i & 63;
        tenc[i] = f2bf(cosf(relt[e] * tw[j] + tb[j]));
    }
}

__global__ void build_h0_k(const float* __restrict__ x,
                           const float* __restrict__ mem,
                           unsigned short* __restrict__ h0, int N) {
    int i = blockIdx.x * blockDim.x + threadIdx.x;
    if (i < N * 192) {
        int n = i / 192, c = i % 192;
        float v = (c < 128) ? x[(size_t)n * 128 + c] : mem[(size_t)n * 64 + (c - 128)];
        h0[i] = f2bf(v);
    }
}

__global__ void cvt_bf16_k(const float* __restrict__ src,
                           unsigned short* __restrict__ dst, int n) {
    int i = blockIdx.x * blockDim.x + threadIdx.x;
    if (i < n) dst[i] = f2bf(src[i]);
}

// Pack a K x 128 f32 matrix into bf16 layout [kChunk][colOff+c][32 klocal]
__global__ void pack_w_k(const float* __restrict__ src, unsigned short* __restrict__ dst,
                         int K, int NOUT, int colOff) {
    int i = blockIdx.x * blockDim.x + threadIdx.x;
    if (i < K * 128) {
        int kl = i & 31;
        int c  = (i >> 5) & 127;
        int kc = i >> 12;                         // 128*32 = 4096 per k-chunk
        dst[((size_t)kc * NOUT + colOff + c) * 32 + kl] =
            f2bf(src[(size_t)(kc * 32 + kl) * 128 + c]);
    }
}

__global__ void pack_b_k(const float* __restrict__ src, float* __restrict__ dst, int n) {
    int i = blockIdx.x * blockDim.x + threadIdx.x;
    if (i < n) dst[i] = src[i];
}

// hnew = skip branch of qkvs; smax = -inf; ssum = 0
__global__ void init_node_k(const float* __restrict__ qkvs,
                            float* __restrict__ hnew,
                            float* __restrict__ smax,
                            float* __restrict__ ssum, int N) {
    int i = blockIdx.x * blockDim.x + threadIdx.x;
    if (i < N * 128) {
        int n = i >> 7, c = i & 127;
        hnew[i] = qkvs[(size_t)n * 512 + 384 + c];
        if (c == 0) { smax[n] = __uint_as_float(0xFF800000u); ssum[n] = 0.f; }
    }
}

__device__ __forceinline__ void atomicMaxF(float* addr, float v) {
    if (v >= 0.f) atomicMax((int*)addr, __float_as_int(v));
    else          atomicMin((unsigned int*)addr, __float_as_uint(v));
}

// wave-per-edge: logits[e] = <q[dst], k[src]+e_emb[e]> / sqrt(128); atomic max
__global__ __launch_bounds__(256)
void edge_logits_k(const long long* __restrict__ ei,
                   const float* __restrict__ qkvs,
                   const float* __restrict__ e_emb,
                   float* __restrict__ logits,
                   float* __restrict__ smax, int E) {
    const int wave = threadIdx.x >> 5, lane = threadIdx.x & 31;
    const int e = blockIdx.x * 8 + wave;
    if (e >= E) return;
    const long long src = ei[e];
    const long long dst = ei[(size_t)E + e];
    const float4 q = *(const float4*)(qkvs + (size_t)dst * 512 + lane * 4);
    const float4 k = *(const float4*)(qkvs + (size_t)src * 512 + 128 + lane * 4);
    const float4 m = *(const float4*)(e_emb + (size_t)e * 128 + lane * 4);
    float p = q.x * (k.x + m.x) + q.y * (k.y + m.y) +
              q.z * (k.z + m.z) + q.w * (k.w + m.w);
#pragma unroll
    for (int off = 16; off > 0; off >>= 1) p += __shfl_xor(p, off, 32);
    if (lane == 0) {
        float lg = p * 0.08838834764831845f;   // 1/sqrt(128)
        logits[e] = lg;
        atomicMaxF(&smax[dst], lg);
    }
}

__global__ void edge_exp_k(const long long* __restrict__ ei,
                           float* __restrict__ logits,
                           const float* __restrict__ smax,
                           float* __restrict__ ssum, int E) {
    int e = blockIdx.x * blockDim.x + threadIdx.x;
    if (e < E) {
        long long dst = ei[(size_t)E + e];
        float a = expf(logits[e] - smax[dst]);
        logits[e] = a;
        atomicAdd(&ssum[dst], a);
    }
}

// wave-per-edge: hnew[dst] += alpha * (v[src] + e_emb[e])
__global__ __launch_bounds__(256)
void edge_agg_k(const long long* __restrict__ ei,
                const float* __restrict__ qkvs,
                const float* __restrict__ e_emb,
                const float* __restrict__ logits,
                const float* __restrict__ ssum,
                float* __restrict__ hnew, int E) {
    const int wave = threadIdx.x >> 5, lane = threadIdx.x & 31;
    const int e = blockIdx.x * 8 + wave;
    if (e >= E) return;
    const long long src = ei[e];
    const long long dst = ei[(size_t)E + e];
    const float alpha = logits[e] / (ssum[dst] + 1e-16f);
    const float4 v = *(const float4*)(qkvs + (size_t)src * 512 + 256 + lane * 4);
    const float4 m = *(const float4*)(e_emb + (size_t)e * 128 + lane * 4);
    float* hp = hnew + (size_t)dst * 128 + lane * 4;
    atomicAdd(hp + 0, alpha * (v.x + m.x));
    atomicAdd(hp + 1, alpha * (v.y + m.y));
    atomicAdd(hp + 2, alpha * (v.z + m.z));
    atomicAdd(hp + 3, alpha * (v.w + m.w));
}

// ---------------------------------------------------------------------------
// Host orchestration
// ---------------------------------------------------------------------------
extern "C" void kernel_launch(void* const* d_in, const int* in_sizes, int n_in,
                              void* d_out, int out_size, void* d_ws, size_t ws_size,
                              hipStream_t stream) {
    const float*     x     = (const float*)d_in[0];
    const float*     mem   = (const float*)d_in[1];
    const long long* ei    = (const long long*)d_in[2];
    const float*     eattr = (const float*)d_in[3];
    const float*     relt  = (const float*)d_in[4];
    const int N = in_sizes[0] / 128;
    const int E = in_sizes[4];

    // --- resolve param flattening order (insertion vs tree-sorted keys) ---
    int i_tw, i_tb, i_Wout, i_bout;
    int iW[4][5], ib[4][5];                 // order: q,k,v,e,skip
    if (in_sizes[5] == 64) {                // insertion: time_w,time_b,layers,Wout,bout
        i_tw = 5; i_tb = 6;
        for (int l = 0; l < 4; ++l) {
            int b = 7 + l * 10;             // Wq,bq,Wk,bk,Wv,bv,We,be,Wskip,bskip
            iW[l][0] = b + 0; ib[l][0] = b + 1;
            iW[l][1] = b + 2; ib[l][1] = b + 3;
            iW[l][2] = b + 4; ib[l][2] = b + 5;
            iW[l][3] = b + 6; ib[l][3] = b + 7;
            iW[l][4] = b + 8; ib[l][4] = b + 9;
        }
        i_Wout = 47; i_bout = 48;
    } else {                                // sorted: Wout,bout,layers,time_b,time_w
        i_Wout = 5; i_bout = 6;
        for (int l = 0; l < 4; ++l) {
            int b = 7 + l * 10;             // We,Wk,Wq,Wskip,Wv,be,bk,bq,bskip,bv
            iW[l][3] = b + 0; iW[l][1] = b + 1; iW[l][0] = b + 2;
            iW[l][4] = b + 3; iW[l][2] = b + 4;
            ib[l][3] = b + 5; ib[l][1] = b + 6; ib[l][0] = b + 7;
            ib[l][4] = b + 8; ib[l][2] = b + 9;
        }
        i_tb = 47; i_tw = 48;
    }

    // --- workspace layout ---
    char* p = (char*)d_ws;
    auto alloc = [&](size_t bytes) -> char* {
        char* r = p; p += (bytes + 255) & ~(size_t)255; return r;
    };
    unsigned short* tenc   = (unsigned short*)alloc((size_t)E * 64  * 2);  // bf16
    unsigned short* h0     = (unsigned short*)alloc((size_t)N * 192 * 2);  // bf16
    unsigned short* eab    = (unsigned short*)alloc((size_t)E * 128 * 2);  // bf16 edge_attr
    unsigned short* hb     = (unsigned short*)alloc((size_t)N * 128 * 2);  // bf16 h
    float* e_emb  = (float*)alloc((size_t)E * 128 * 4);
    float* qkvs   = (float*)alloc((size_t)N * 512 * 4);
    float* hnew   = (float*)alloc((size_t)N * 128 * 4);
    float* logits = (float*)alloc((size_t)E * 4);
    float* smax   = (float*)alloc((size_t)N * 4);
    float* ssum   = (float*)alloc((size_t)N * 4);
    float* biasP  = (float*)alloc((size_t)(4 * 1024 + 128) * 4);
    unsigned short* Wqk[4];
    for (int l = 0; l < 4; ++l) {
        int K = (l == 0) ? 192 : 320;
        Wqk[l] = (unsigned short*)alloc((size_t)K * 512 * 2);
    }
    unsigned short* WeP[4];
    for (int l = 0; l < 4; ++l) WeP[l] = (unsigned short*)alloc((size_t)192 * 128 * 2);
    unsigned short* WoutP = (unsigned short*)alloc((size_t)128 * 128 * 2);

    auto g1 = [](int n) { return (n + 255) / 256; };

    // --- pack weights/biases to bf16 (swizzled [kChunk][col][32]) ---
    for (int l = 0; l < 4; ++l) {
        const int K = (l == 0) ? 192 : 320;
        for (int j = 0; j < 5; ++j) {
            const float* Wsrc = (const float*)d_in[iW[l][j]];
            const float* bsrc = (const float*)d_in[ib[l][j]];
            if (j == 3) { // We -> separate 192x128 matrix
                pack_w_k<<<g1(192 * 128), 256, 0, stream>>>(Wsrc, WeP[l], 192, 128, 0);
                pack_b_k<<<g1(128), 256, 0, stream>>>(bsrc, biasP + l * 1024 + 512, 128);
            } else {      // q,k,v,skip -> fused K x 512
                int colOff = (j < 3) ? j * 128 : 384;
                pack_w_k<<<g1(K * 128), 256, 0, stream>>>(Wsrc, Wqk[l], K, 512, colOff);
                pack_b_k<<<g1(128), 256, 0, stream>>>(bsrc, biasP + l * 1024 + colOff, 128);
            }
        }
    }
    pack_w_k<<<g1(128 * 128), 256, 0, stream>>>((const float*)d_in[i_Wout], WoutP, 128, 128, 0);
    pack_b_k<<<g1(128), 256, 0, stream>>>((const float*)d_in[i_bout], biasP + 4096, 128);

    // --- prep: bf16 activations ---
    time_encode_k<<<g1(E * 64), 256, 0, stream>>>(relt, (const float*)d_in[i_tw],
                                                  (const float*)d_in[i_tb], tenc, E);
    build_h0_k<<<g1(N * 192), 256, 0, stream>>>(x, mem, h0, N);
    cvt_bf16_k<<<g1(E * 128), 256, 0, stream>>>(eattr, eab, E * 128);

    const int gbE = (E + 31) / 32, gbN = (N + 31) / 32;
    const int geW = (E + 7) / 8;

    // --- 4 transformer-conv blocks ---
    for (int l = 0; l < 4; ++l) {
        gemm_bf16<2><<<gbE, 256, 0, stream>>>(tenc, 64, eab, 128,
                                              WeP[l], biasP + l * 1024 + 512, e_emb, E);
        if (l == 0)
            gemm_bf16<8><<<gbN, 256, 0, stream>>>(h0, 192, (const unsigned short*)nullptr, 0,
                                                  Wqk[0], biasP + 0, qkvs, N);
        else {
            cvt_bf16_k<<<g1(N * 128), 256, 0, stream>>>(hnew, hb, N * 128);
            gemm_bf16<8><<<gbN, 256, 0, stream>>>(hb, 128, h0, 192,
                                                  Wqk[l], biasP + l * 1024, qkvs, N);
        }
        init_node_k<<<g1(N * 128), 256, 0, stream>>>(qkvs, hnew, smax, ssum, N);
        edge_logits_k<<<geW, 256, 0, stream>>>(ei, qkvs, e_emb, logits, smax, E);
        edge_exp_k<<<g1(E), 256, 0, stream>>>(ei, logits, smax, ssum, E);
        edge_agg_k<<<geW, 256, 0, stream>>>(ei, qkvs, e_emb, logits, ssum, hnew, E);
    }

    // --- output projection: d_out = h @ Wout + bout ---
    cvt_bf16_k<<<g1(N * 128), 256, 0, stream>>>(hnew, hb, N * 128);
    gemm_bf16<2><<<gbN, 256, 0, stream>>>(hb, 128, (const unsigned short*)nullptr, 0,
                                          WoutP, biasP + 4096, (float*)d_out, N);
}
